// V7AstrometryMatcher_48447231098967
// MI455X (gfx1250) — matrix operand order repaired and checked by reading, not compiled, not run on hardware
//
#include <hip/hip_runtime.h>
#include <hip/hip_bf16.h>
#include <stdint.h>
#include <math.h>

// ---------------------------------------------------------------------------
// MI455X (gfx1250) implementation of V7AstrometryMatcher forward.
// Memory-bound core: 268MB fp32 input reads ~= 11.5us @ 23.3 TB/s.
// Projection GEMM + MLP head use v_wmma_f32_16x16x32_f16 (f32 accumulate).
// Intermediate normalized features stored f16 (halves 2nd-pass HBM traffic).
// Correlation uses packed f16 FMA (v_pk_fma_f16) so its VALU cost sits below
// the HBM roofline (was the bottleneck as scalar f32 fma + cvt).
// ---------------------------------------------------------------------------

typedef __attribute__((ext_vector_type(16))) _Float16 v16h;
typedef __attribute__((ext_vector_type(8)))  float    v8f;
typedef __attribute__((ext_vector_type(2)))  _Float16 h2;

#define B_N   128
#define C_N   64
#define HW_N  4096
#define KK_N  49

__device__ __forceinline__ v8f wmma_f16(v16h a, v16h b, v8f c) {
  // 8 args: (neg_a, A, neg_b, B, c_mod, C, reuse_a, reuse_b)
  return __builtin_amdgcn_wmma_f32_16x16x32_f16(false, a, false, b, (short)0, c,
                                                false, false);
}

// A-matrix 16x32 f16 fragment gather (ISA 7.12.2): lane L holds row M=L%16,
// K = kstep*32 + (L>=16?8:0) + {2v,2v+1} for v<4, +16 for v>=4.
__device__ __forceinline__ v16h load_a_frag(const _Float16* base, int m,
                                            int lane, int kstep, int stride) {
  const _Float16* row = base + m * stride;
  const int kb = kstep * 32 + ((lane & 16) ? 8 : 0);
  v16h a;
#pragma unroll
  for (int v = 0; v < 8; ++v) {
    int k = kb + ((v < 4) ? (2 * v) : (16 + 2 * (v - 4)));
    a[2 * v]     = row[k];
    a[2 * v + 1] = row[k + 1];
  }
  return a;
}

// B-matrix 32x16 f16 fragment: lane L holds column N, K = kstep*32 +
// (L>=16?8:0) + i for i<8, +16 for i>=8 (matches A).  LDS layout [n][k].
__device__ __forceinline__ v16h load_b_frag(const _Float16* base, int n,
                                            int lane, int kstep, int stride) {
  const _Float16* p = base + n * stride + kstep * 32 + ((lane & 16) ? 8 : 0);
  v16h b;
#pragma unroll
  for (int i = 0; i < 16; ++i) b[i] = p[(i < 8) ? i : (i + 8)];
  return b;
}

// ---------------------------------------------------------------------------
// k0: gauss window (normalized) + zero accumulators (graph-replay safe).
// ---------------------------------------------------------------------------
__global__ void k0_init(float* __restrict__ gauss, float* __restrict__ cost,
                        float* __restrict__ rpool, float* __restrict__ vpool) {
  __shared__ float red[256];
  const int t = threadIdx.x;
  float vals[16];
  float lsum = 0.f;
#pragma unroll
  for (int j = 0; j < 16; ++j) {
    int i = t * 16 + j;
    int iy = i >> 6, ix = i & 63;
    float y = -1.f + (2.f / 63.f) * (float)iy;
    float x = -1.f + (2.f / 63.f) * (float)ix;
    float g = expf(-2.f * (y * y + x * x));   // sigma = 0.5
    vals[j] = g;
    lsum += g;
  }
  red[t] = lsum;
  __syncthreads();
  for (int s = 128; s > 0; s >>= 1) {
    if (t < s) red[t] += red[t + s];
    __syncthreads();
  }
  const float inv = 1.f / red[0];
#pragma unroll
  for (int j = 0; j < 16; ++j) gauss[t * 16 + j] = vals[j] * inv;
  for (int i = t; i < B_N * KK_N; i += 256) cost[i] = 0.f;
  for (int i = t; i < B_N * C_N; i += 256) { rpool[i] = 0.f; vpool[i] = 0.f; }
}

// ---------------------------------------------------------------------------
// k1: dual projection GEMM (WMMA f16) + L2-normalize + energy + gauss pooling.
// One block = one batch x 128-pixel column tile. M=64 (4 tiles), K=64 (2 steps).
// ---------------------------------------------------------------------------
__global__ __launch_bounds__(256) void k1_proj(
    const float* __restrict__ xr, const float* __restrict__ xv,
    const float* __restrict__ wr, const float* __restrict__ wv_,
    const float* __restrict__ gauss,
    _Float16* __restrict__ rn, _Float16* __restrict__ vn,
    float* __restrict__ energy,
    float* __restrict__ rpool, float* __restrict__ vpool) {
  __shared__ __align__(32) _Float16 Ar[4096];   // pre-swizzled Wr fragments
  __shared__ __align__(32) _Float16 Av[4096];   // pre-swizzled Wv fragments
  __shared__ __align__(32) _Float16 Br[8192];   // X_r tile [px][c]
  __shared__ __align__(32) _Float16 Bv[8192];   // X_v tile [px][c]
  __shared__ float pR[64], pV[64];

  const int t = threadIdx.x;
  const int b = blockIdx.y;
  const int pbase = blockIdx.x * 128;
  const int lane = t & 31;
  const int wid = t >> 5;

  if (t < 64) { pR[t] = 0.f; pV[t] = 0.f; }

  // Stage weights directly in A-fragment lane order: slot = (mt*2+ks)*32+lane.
  {
    const int s = t;
    const int mt = s >> 6;
    const int ks = (s >> 5) & 1;
    const int ln = s & 31;
    const int m = mt * 16 + (ln & 15);
    const int kb = ks * 32 + ((ln & 16) ? 8 : 0);
#pragma unroll
    for (int e = 0; e < 16; ++e) {
      int v = e >> 1, h = e & 1;
      int k = kb + ((v < 4) ? (2 * v) : (16 + 2 * (v - 4))) + h;
      Ar[s * 16 + e] = (_Float16)wr[m * 64 + k];
      Av[s * 16 + e] = (_Float16)wv_[m * 64 + k];
    }
  }
  __syncthreads();

  // Stage B tiles (coalesced fp32 loads, f16 LDS) + fused gauss pooling.
  {
    const int i = t & 127;          // pixel in tile (fixed per thread)
    const int crow = t >> 7;        // 0 or 1 (uniform per wave)
    const int p = pbase + i;
    const float g = gauss[p];
    const size_t gb = (size_t)b * C_N * HW_N;
    for (int j = 0; j < 32; ++j) {
      const int c = crow + 2 * j;   // uniform across the wave
      if (j + 1 < 32) {             // streaming hint -> global_prefetch_b8
        __builtin_prefetch(&xr[gb + (size_t)(c + 2) * HW_N + p], 0, 1);
        __builtin_prefetch(&xv[gb + (size_t)(c + 2) * HW_N + p], 0, 1);
      }
      float fr = xr[gb + (size_t)c * HW_N + p];
      float fv = xv[gb + (size_t)c * HW_N + p];
      Br[i * 64 + c] = (_Float16)fr;
      Bv[i * 64 + c] = (_Float16)fv;
      float sr = fr * g, sv = fv * g;
#pragma unroll
      for (int o = 16; o > 0; o >>= 1) {
        sr += __shfl_xor(sr, o, 32);
        sv += __shfl_xor(sv, o, 32);
      }
      if (lane == 0) { atomicAdd(&pR[c], sr); atomicAdd(&pV[c], sv); }
    }
  }
  __syncthreads();

  // WMMA compute: wave wid owns 16-pixel N subtile.
  const int px0 = wid * 16;
  v8f accr[4], accv[4];
#pragma unroll
  for (int mt = 0; mt < 4; ++mt)
#pragma unroll
    for (int r = 0; r < 8; ++r) { accr[mt][r] = 0.f; accv[mt][r] = 0.f; }

#pragma unroll
  for (int ks = 0; ks < 2; ++ks) {
    v16h br = load_b_frag(Br, px0 + (lane & 15), lane, ks, 64);
    v16h bv = load_b_frag(Bv, px0 + (lane & 15), lane, ks, 64);
#pragma unroll
    for (int mt = 0; mt < 4; ++mt) {
      v16h ar = *(const v16h*)&Ar[((mt * 2 + ks) * 32 + lane) * 16];
      accr[mt] = wmma_f16(ar, br, accr[mt]);
      v16h av = *(const v16h*)&Av[((mt * 2 + ks) * 32 + lane) * 16];
      accv[mt] = wmma_f16(av, bv, accv[mt]);
    }
  }

  // Per-pixel channel L2 norm: lanes L and L^16 jointly hold all 64 channels.
  float ssr = 0.f, ssv = 0.f;
#pragma unroll
  for (int mt = 0; mt < 4; ++mt)
#pragma unroll
    for (int r = 0; r < 8; ++r) {
      ssr += accr[mt][r] * accr[mt][r];
      ssv += accv[mt][r] * accv[mt][r];
    }
  ssr += __shfl_xor(ssr, 16, 32);
  ssv += __shfl_xor(ssv, 16, 32);
  const float inr = 1.f / fmaxf(sqrtf(ssr), 1e-6f);
  const float inv = 1.f / fmaxf(sqrtf(ssv), 1e-6f);
#pragma unroll
  for (int mt = 0; mt < 4; ++mt)
#pragma unroll
    for (int r = 0; r < 8; ++r) { accr[mt][r] *= inr; accv[mt][r] *= inv; }
  if (lane < 16)
    energy[(size_t)b * HW_N + pbase + px0 + lane] = ssr * inr * inr;

  __syncthreads();
  // Restage normalized f16 outputs into Br/Bv ([px][c]) for coalesced stores.
  {
    const int px = px0 + (lane & 15);
    const int chi = (lane & 16) ? 8 : 0;
#pragma unroll
    for (int mt = 0; mt < 4; ++mt)
#pragma unroll
      for (int r = 0; r < 8; ++r) {
        const int c = 16 * mt + chi + r;
        Br[px * 64 + c] = (_Float16)accr[mt][r];
        Bv[px * 64 + c] = (_Float16)accv[mt][r];
      }
  }
  __syncthreads();
  {
    const uint32_t* s1 = (const uint32_t*)Br;
    const uint32_t* s2 = (const uint32_t*)Bv;
    uint32_t* d1 = (uint32_t*)(rn + ((size_t)b * HW_N + pbase) * 64);
    uint32_t* d2 = (uint32_t*)(vn + ((size_t)b * HW_N + pbase) * 64);
#pragma unroll
    for (int j = 0; j < 16; ++j) {
      d1[t + 256 * j] = s1[t + 256 * j];
      d2[t + 256 * j] = s2[t + 256 * j];
    }
  }
  if (t < 64) {
    atomicAdd(&rpool[b * 64 + t], pR[t]);
    atomicAdd(&vpool[b * 64 + t], pV[t]);
  }
}

// ---------------------------------------------------------------------------
// k2: spatial-weight denominator per batch.
// ---------------------------------------------------------------------------
__global__ void k2_denom(const float* __restrict__ energy,
                         const float* __restrict__ gauss,
                         float* __restrict__ denom) {
  __shared__ float red[256];
  const int b = blockIdx.x, t = threadIdx.x;
  float s = 0.f;
  for (int i = t; i < HW_N; i += 256) s += energy[(size_t)b * HW_N + i] * gauss[i];
  red[t] = s;
  __syncthreads();
  for (int k = 128; k > 0; k >>= 1) {
    if (t < k) red[t] += red[t + k];
    __syncthreads();
  }
  if (t == 0) denom[b] = red[0] + 1e-8f;
}

// ---------------------------------------------------------------------------
// k3: 49-shift correlation from LDS tiles (16x16 tile, halo 3, replicate pad).
// Dynamic LDS: vn halo 22*22*64 f16 (60.5KB) + rn 256*64 f16 (32KB) + 49 f32.
// Inner dot products in packed f16 (v_pk_fma_f16): operands are unit-norm
// features, |term| <= 1, |sum| <= 64 -> comfortably inside f16 range.
// ---------------------------------------------------------------------------
__global__ __launch_bounds__(256) void k3_corr(
    const _Float16* __restrict__ rn, const _Float16* __restrict__ vn,
    const float* __restrict__ energy, const float* __restrict__ gauss,
    const float* __restrict__ denom, float* __restrict__ cost) {
  extern __shared__ char smem[];
  _Float16* vnT = (_Float16*)smem;          // [22][22][64]
  _Float16* rnT = vnT + 22 * 22 * 64;       // [256][64]
  float* costL = (float*)(rnT + 256 * 64);  // [49]

  const int t = threadIdx.x;
  const int b = blockIdx.y;
  const int ty = (blockIdx.x >> 2) * 16;
  const int tx = (blockIdx.x & 3) * 16;
  const size_t gdw = (size_t)b * HW_N * 32;  // dword offset of batch

  if (t < KK_N) costL[t] = 0.f;

  {  // rn tile (dword copies, 128B per pixel row)
    const uint32_t* g = (const uint32_t*)rn;
    uint32_t* d = (uint32_t*)rnT;
    for (int q = t; q < 256 * 32; q += 256) {
      const int pix = q >> 5, w = q & 31;
      const int py = pix >> 4, px = pix & 15;
      d[q] = g[gdw + (size_t)((ty + py) * 64 + tx + px) * 32 + w];
    }
  }
  {  // vn halo with replicate-pad clamp
    const uint32_t* g = (const uint32_t*)vn;
    uint32_t* d = (uint32_t*)vnT;
    for (int q = t; q < 484 * 32; q += 256) {
      const int pix = q >> 5, w = q & 31;
      const int hy = pix / 22, hx = pix % 22;
      int gy = ty + hy - 3; gy = gy < 0 ? 0 : (gy > 63 ? 63 : gy);
      int gx = tx + hx - 3; gx = gx < 0 ? 0 : (gx > 63 ? 63 : gx);
      d[q] = g[gdw + (size_t)(gy * 64 + gx) * 32 + w];
    }
  }
  __syncthreads();

  const int py = t >> 4, px = t & 15;
  const int pg = (ty + py) * 64 + tx + px;
  const float sw = energy[(size_t)b * HW_N + pg] * gauss[pg] / denom[b];

  // Cache this pixel's 64 channels as 32 packed f16 pairs (registers).
  const h2* rrow = (const h2*)&rnT[t * 64];
  h2 rrh[32];
#pragma unroll
  for (int c = 0; c < 32; ++c) rrh[c] = rrow[c];

  for (int dy = 0; dy < 7; ++dy) {
    for (int dx = 0; dx < 7; ++dx) {
      const h2* vrow = (const h2*)&vnT[((py + dy) * 22 + (px + dx)) * 64];
      h2 a0; a0[0] = (_Float16)0.f; a0[1] = (_Float16)0.f;
      h2 a1 = a0;
#pragma unroll
      for (int c = 0; c < 32; c += 2) {       // v_pk_fma_f16 x2 per iter
        a0 = rrh[c] * vrow[c] + a0;
        a1 = rrh[c + 1] * vrow[c + 1] + a1;
      }
      float s = ((float)a0[0] + (float)a0[1]) + ((float)a1[0] + (float)a1[1]);
      s *= sw;
#pragma unroll
      for (int o = 16; o > 0; o >>= 1) s += __shfl_xor(s, o, 32);
      if ((t & 31) == 0) atomicAdd(&costL[dy * 7 + dx], s);
    }
  }
  __syncthreads();
  if (t < KK_N) atomicAdd(&cost[b * KK_N + t], costL[t]);
}

// ---------------------------------------------------------------------------
// k4: softmax head + 2-layer WMMA MLP (M=128 batches) + final affine outputs.
// Dynamic LDS: feat[128][224] + wT[128][224] + h1[128][128] + h2[128][128] f16.
// ---------------------------------------------------------------------------
__global__ __launch_bounds__(256) void k4_head(
    const float* __restrict__ cost, const float* __restrict__ rpool,
    const float* __restrict__ vpool, const float* __restrict__ log_temp,
    const float* __restrict__ band_emb, const int* __restrict__ band_idx,
    const float* __restrict__ w1, const float* __restrict__ b1,
    const float* __restrict__ w2, const float* __restrict__ b2,
    const float* __restrict__ w3, const float* __restrict__ b3,
    const float* __restrict__ p2s, float* __restrict__ out, int ostride) {
  extern __shared__ char smem[];
  _Float16* featL = (_Float16*)smem;          // [128][224]
  _Float16* wL = featL + 128 * 224;           // [n][k] stride 224
  _Float16* h1L = wL + 128 * 224;             // [m][n] stride 128
  _Float16* h2L = h1L + 128 * 128;            // [m][n] stride 128
  __shared__ float cdx[128], cdy[128], cf[128];

  const int t = threadIdx.x;
  const int lane = t & 31;
  const int wid = t >> 5;

  // Phase A: per-batch softmax over 49 logits + feature row assembly.
  if (t < 128) {
    const int b = t;
    const float temp = fmaxf(expf(log_temp[0]), 1e-3f);
    const float sc = 1.f / (8.f * temp);  // sqrt(C)=8 folded with temp
    float l[KK_N];
    float m = -1e30f;
#pragma unroll
    for (int d = 0; d < KK_N; ++d) {
      l[d] = cost[b * KK_N + d] * sc;
      m = fmaxf(m, l[d]);
    }
    float Z = 0.f, sdx = 0.f, sdy = 0.f;
#pragma unroll
    for (int d = 0; d < KK_N; ++d) {
      float e = expf(l[d] - m);
      Z += e;
      sdx += e * (float)((d % 7) - 3);
      sdy += e * (float)((d / 7) - 3);
    }
    const float iZ = 1.f / Z;
    cdx[b] = sdx * iZ;
    cdy[b] = sdy * iZ;
    cf[b] = iZ;  // max prob = exp(0)/Z
    const int bi = band_idx[b];
#pragma unroll
    for (int c = 0; c < 64; ++c) {
      float r = rpool[b * 64 + c], v = vpool[b * 64 + c];
      featL[b * 224 + c] = (_Float16)r;
      featL[b * 224 + 64 + c] = (_Float16)v;
      featL[b * 224 + 128 + c] = (_Float16)(r - v);
    }
    featL[b * 224 + 192] = (_Float16)(sdx * iZ);
    featL[b * 224 + 193] = (_Float16)(sdy * iZ);
#pragma unroll
    for (int e = 0; e < 16; ++e)
      featL[b * 224 + 194 + e] = (_Float16)band_emb[bi * 16 + e];
#pragma unroll
    for (int k = 210; k < 224; ++k) featL[b * 224 + k] = (_Float16)0.f;
  }
  // Stage w1^T -> wL[n][k], zero-padded K to 224.
  for (int q = t; q < 128 * 224; q += 256) {
    const int n = q / 224, k = q % 224;
    wL[q] = (_Float16)((k < 210) ? w1[k * 128 + n] : 0.f);
  }
  __syncthreads();

  // Layer 1: h1 = gelu(feat @ w1 + b1), K=224 (7 ksteps).
  {
    const int n = wid * 16 + (lane & 15);
    const float bias = b1[n];
    for (int mt = 0; mt < 8; ++mt) {
      v8f acc;
#pragma unroll
      for (int r = 0; r < 8; ++r) acc[r] = 0.f;
      for (int ks = 0; ks < 7; ++ks) {
        v16h a = load_a_frag(featL, mt * 16 + (lane & 15), lane, ks, 224);
        v16h bf = load_b_frag(wL, n, lane, ks, 224);
        acc = wmma_f16(a, bf, acc);
      }
      const int mb = mt * 16 + ((lane & 16) ? 8 : 0);
#pragma unroll
      for (int r = 0; r < 8; ++r) {
        float x = acc[r] + bias;
        h1L[(mb + r) * 128 + n] =
            (_Float16)(0.5f * x * (1.f + erff(x * 0.70710678118f)));
      }
    }
  }
  __syncthreads();
  // Stage w2^T (reuse wL, stride 224, only k<128 read).
  for (int q = t; q < 128 * 128; q += 256) {
    const int n = q >> 7, k = q & 127;
    wL[n * 224 + k] = (_Float16)w2[k * 128 + n];
  }
  __syncthreads();

  // Layer 2: h2 = gelu(h1 @ w2 + b2), K=128 (4 ksteps).
  {
    const int n = wid * 16 + (lane & 15);
    const float bias = b2[n];
    for (int mt = 0; mt < 8; ++mt) {
      v8f acc;
#pragma unroll
      for (int r = 0; r < 8; ++r) acc[r] = 0.f;
      for (int ks = 0; ks < 4; ++ks) {
        v16h a = load_a_frag(h1L, mt * 16 + (lane & 15), lane, ks, 128);
        v16h bf = load_b_frag(wL, n, lane, ks, 224);
        acc = wmma_f16(a, bf, acc);
      }
      const int mb = mt * 16 + ((lane & 16) ? 8 : 0);
#pragma unroll
      for (int r = 0; r < 8; ++r) {
        float x = acc[r] + bias;
        h2L[(mb + r) * 128 + n] =
            (_Float16)(0.5f * x * (1.f + erff(x * 0.70710678118f)));
      }
    }
  }
  __syncthreads();

  // Phase C: tiny N=3 output layer + affine sky transform.
  if (t < 128) {
    const int b = t;
    float o0 = b3[0], o1 = b3[1], o2 = b3[2];
    for (int k = 0; k < 128; ++k) {
      const float h = (float)h2L[b * 128 + k];
      o0 += h * w3[k * 3 + 0];
      o1 += h * w3[k * 3 + 1];
      o2 += h * w3[k * 3 + 2];
    }
    const float dx = cdx[b] + o0;
    const float dy = cdy[b] + o1;
    const float ls = fminf(fmaxf(o2, -6.f), 3.f);
    const float s0 = p2s[b * 4 + 0] * dx + p2s[b * 4 + 1] * dy;
    const float s1 = p2s[b * 4 + 2] * dx + p2s[b * 4 + 3] * dy;
    float* o = out + (size_t)b * ostride;
    o[0] = dx; o[1] = dy; o[2] = s0; o[3] = s1; o[4] = ls; o[5] = cf[b];
  }
}

// ---------------------------------------------------------------------------
extern "C" void kernel_launch(void* const* d_in, const int* in_sizes, int n_in,
                              void* d_out, int out_size, void* d_ws,
                              size_t ws_size, hipStream_t stream) {
  const float* rubin = (const float*)d_in[0];
  const float* vis = (const float*)d_in[1];
  const float* p2s = (const float*)d_in[2];
  const int* bidx = (const int*)d_in[3];
  const float* wr = (const float*)d_in[4];
  const float* wvp = (const float*)d_in[5];
  const float* ltemp = (const float*)d_in[6];
  const float* bemb = (const float*)d_in[7];
  const float* w1 = (const float*)d_in[8];
  const float* b1 = (const float*)d_in[9];
  const float* w2 = (const float*)d_in[10];
  const float* b2 = (const float*)d_in[11];
  const float* w3 = (const float*)d_in[12];
  const float* b3 = (const float*)d_in[13];

  char* ws = (char*)d_ws;
  size_t off = 0;
  _Float16* rn = (_Float16*)(ws + off); off += (size_t)B_N * HW_N * C_N * 2;
  _Float16* vn = (_Float16*)(ws + off); off += (size_t)B_N * HW_N * C_N * 2;
  float* energy = (float*)(ws + off); off += (size_t)B_N * HW_N * 4;
  float* gauss = (float*)(ws + off); off += HW_N * 4;
  float* denom = (float*)(ws + off); off += B_N * 4;
  float* costb = (float*)(ws + off); off += B_N * KK_N * 4;
  float* rpool = (float*)(ws + off); off += B_N * C_N * 4;
  float* vpool = (float*)(ws + off); off += B_N * C_N * 4;

  k0_init<<<1, 256, 0, stream>>>(gauss, costb, rpool, vpool);
  k1_proj<<<dim3(32, B_N), 256, 0, stream>>>(rubin, vis, wr, wvp, gauss, rn, vn,
                                             energy, rpool, vpool);
  k2_denom<<<B_N, 256, 0, stream>>>(energy, gauss, denom);
  const size_t smem3 = (size_t)(22 * 22 * 64 + 256 * 64) * 2 + KK_N * 4;
  k3_corr<<<dim3(16, B_N), 256, smem3, stream>>>(rn, vn, energy, gauss, denom,
                                                 costb);
  const size_t smem4 = (size_t)(128 * 224 * 2 + 128 * 128 * 2) * 2;
  k4_head<<<1, 256, smem4, stream>>>(costb, rpool, vpool, ltemp, bemb, bidx, w1,
                                     b1, w2, b2, w3, b3, p2s, (float*)d_out,
                                     out_size / B_N);
}
